// AdvancedFractalUnit_7155415515349
// MI455X (gfx1250) — matrix operand order repaired and checked by reading, not compiled, not run on hardware
//
#include <hip/hip_runtime.h>
#include <hip/hip_bf16.h>
#include <math.h>

// ---------------------------------------------------------------------------
// Problem constants (fixed by the reference)
// ---------------------------------------------------------------------------
#define BB   32
#define CIN  128
#define CO   256
#define HH   32
#define WW2  32
#define HW   1024            // H*W
#define NPOS 32768           // B*H*W
#define MEMN 64
#define EPSV 1e-5f

typedef _Float16 h16_t;
typedef __attribute__((ext_vector_type(16))) _Float16 v16h;
typedef __attribute__((ext_vector_type(8)))  _Float16 v8h;
typedef __attribute__((ext_vector_type(8)))  float    v8f;

__device__ __forceinline__ float sigmoidf_(float v) {
    return 1.0f / (1.0f + __expf(-v));
}

// ---------------------------------------------------------------------------
// Templated WMMA GEMM, all-f16 operands (pre-staged), f32 accumulate:
//   out[n*ldo + m] = act( alpha * sum_k X[n,k] * A[m,k] + bias[m] )
//
// A: row-major f16 (lda). 3x3 conv weights are pre-repacked so k = tap*CINv+c.
// TAPS : 1 = plain GEMM, 9 = implicit 3x3 conv (CINv = 1<<LCIN multiple of 32
//        so each 32-wide K tile lies in one tap -> tile-uniform (dh,dw)).
// ACT  : 0 none, 1 relu, 2 sigmoid.  HASB: bias.  HASX2: concat K at ksplit.
// OUTM : 0 f32 only, 1 f16 only, 2 both.
// Wave tile (MI*16)x(NI*16); block (WM*MI*16)x(WN*NI*16); 256 threads.
// Exact M/N coverage by construction -> no predication, EXEC all-ones.
// ---------------------------------------------------------------------------
template<int TAPS, int LCIN, int ACT, int HASB, int HASX2, int HASALPHA,
         int OUTM, int WM, int WN, int MI, int NI>
__global__ void __launch_bounds__(256)
wmma_gemm(float* __restrict__ out, h16_t* __restrict__ out16,
          const h16_t* __restrict__ x1, const h16_t* __restrict__ x2,
          const h16_t* __restrict__ wA, const float* __restrict__ bias,
          int Ktot, int ldo, int ld1, int ld2, int ksplit, int lda, float alpha)
{
    constexpr int CINv = 1 << LCIN;
    const int lane  = threadIdx.x & 31;
    const int wave  = threadIdx.x >> 5;
    const int lhalf = lane >> 4;          // 0 / 1
    const int lmod  = lane & 15;
    const int waveM = wave / WN;
    const int waveN = wave % WN;
    const int mBase = blockIdx.x * (WM * MI * 16) + waveM * (MI * 16);
    const int nBase = blockIdx.y * (WN * NI * 16) + waveN * (NI * 16);

    v8f acc[MI][NI] = {};

    for (int k0 = 0; k0 < Ktot; k0 += 32) {
        int dh = 0, dw = 0, c0 = k0;
        if (TAPS == 9) {
            const int tap = k0 >> LCIN;
            c0 = k0 & (CINv - 1);
            dh = tap / 3 - 1;
            dw = tap - (tap / 3) * 3 - 1;
        }

        // ---- A fragments (16x32 f16): element i -> K = i + 16*lhalf
        //      -> two contiguous 8-half runs per lane (2x 16B loads).
        v16h afrag[MI];
        #pragma unroll
        for (int mi = 0; mi < MI; ++mi) {
            const int m = mBase + mi * 16 + lmod;
            const h16_t* ap = wA + (size_t)m * lda + k0 + 16 * lhalf;
            const v8h r0 = *reinterpret_cast<const v8h*>(ap);
            const v8h r1 = *reinterpret_cast<const v8h*>(ap + 8);
            afrag[mi] = __builtin_shufflevector(r0, r1,
                0, 1, 2, 3, 4, 5, 6, 7, 8, 9, 10, 11, 12, 13, 14, 15);
        }

        // ---- B fragments (32x16 f16): element i -> K = i + 16*lhalf
        //      -> 16 contiguous halfs per lane (one 32B vector load).
        v16h bfrag[NI];
        #pragma unroll
        for (int ni = 0; ni < NI; ++ni) {
            const int n = nBase + ni * 16 + lmod;
            const h16_t* src;
            bool valid = true;
            if (TAPS == 9) {
                const int b  = n >> 10;           // HW = 1024
                const int hw = n & 1023;
                const int h  = hw >> 5;
                const int w  = hw & 31;
                const int hs = h + dh, ws = w + dw;
                valid = (hs >= 0) & (hs < HH) & (ws >= 0) & (ws < WW2);
                const size_t off =
                    ((size_t)((b * HH + (valid ? hs : h)) * WW2 +
                              (valid ? ws : w)) << LCIN) + c0;
                src = x1 + off;                   // always a safe address
            } else if (HASX2 && k0 >= ksplit) {
                src = x2 + (size_t)n * ld2 + (k0 - ksplit);
            } else {
                src = x1 + (size_t)n * ld1 + k0;
            }
            const v16h ldv = *reinterpret_cast<const v16h*>(src + 16 * lhalf);
            if (TAPS == 9) {
                #pragma unroll
                for (int i = 0; i < 16; ++i)
                    bfrag[ni][i] = valid ? ldv[i] : (h16_t)0.0f;
            } else {
                bfrag[ni] = ldv;
            }
        }

        // ---- WMMA (block-uniform control flow: EXEC all-ones)
        #pragma unroll
        for (int mi = 0; mi < MI; ++mi)
            #pragma unroll
            for (int ni = 0; ni < NI; ++ni)
                acc[mi][ni] = __builtin_amdgcn_wmma_f32_16x16x32_f16(
                    false, afrag[mi], false, bfrag[ni],
                    (short)0, acc[mi][ni], false, false);
    }

    // ---- Epilogue. D layout: element r -> M = r + 8*lhalf, N = lmod.
    //      m contiguous over r -> 2x float4 (f32) / 1x v8h (f16) per tile.
    #pragma unroll
    for (int mi = 0; mi < MI; ++mi) {
        const int mRow = mBase + mi * 16 + 8 * lhalf;
        #pragma unroll
        for (int ni = 0; ni < NI; ++ni) {
            const int n = nBase + ni * 16 + lmod;
            float v[8];
            #pragma unroll
            for (int r = 0; r < 8; ++r) {
                float t = acc[mi][ni][r];
                if (HASALPHA) t *= alpha;
                if (HASB)     t += bias[mRow + r];
                if (ACT == 1)      t = fmaxf(t, 0.0f);
                else if (ACT == 2) t = sigmoidf_(t);
                v[r] = t;
            }
            if (OUTM != 1) {
                float* op = out + (size_t)n * ldo + mRow;
                *reinterpret_cast<float4*>(op)     = make_float4(v[0], v[1], v[2], v[3]);
                *reinterpret_cast<float4*>(op + 4) = make_float4(v[4], v[5], v[6], v[7]);
            }
            if (OUTM != 0) {
                v8h hv;
                #pragma unroll
                for (int r = 0; r < 8; ++r) hv[r] = (h16_t)v[r];
                *reinterpret_cast<v8h*>(out16 + (size_t)n * ldo + mRow) = hv;
            }
        }
    }
}

// ---------------------------------------------------------------------------
// Staging kernels: f32 -> f16 conversions / repacks (run once, tiny cost)
// ---------------------------------------------------------------------------
__global__ void cvtf16(const float* __restrict__ in, h16_t* __restrict__ o) {
    const int idx = blockIdx.x * 256 + threadIdx.x;
    o[idx] = (h16_t)in[idx];
}

// conv OIHW -> row-major A[m][k], k = tap*CINv + c, f16.  block 128 threads,
// grid (9*CINv/128, M)
__global__ void repack_w3x3(const float* __restrict__ in, h16_t* __restrict__ o,
                            int lcin) {
    const int kidx = blockIdx.x * 128 + threadIdx.x;
    const int m    = blockIdx.y;
    const int cinv = 1 << lcin;
    const int tap  = kidx >> lcin;
    const int c    = kidx & (cinv - 1);
    o[(size_t)m * (9 << lcin) + kidx] = (h16_t)in[((m << lcin) + c) * 9 + tap];
}

// mem (MEMN x CO) -> memT[m][k] = mem[k*CO + m], f16
__global__ void repack_memT(const float* __restrict__ in, h16_t* __restrict__ o) {
    const int idx = blockIdx.x * 256 + threadIdx.x;   // CO*MEMN
    const int m = idx >> 6, k = idx & 63;
    o[idx] = (h16_t)in[k * CO + m];
}

// NCHW f32 -> position-major NHWC f16
__global__ void nchw_to_nhwc16(const float* __restrict__ x, h16_t* __restrict__ xT) {
    const int idx = blockIdx.x * 256 + threadIdx.x;     // over B*CIN*HW
    const int b  = idx >> 17;                           // CIN*HW = 131072
    const int c  = (idx >> 10) & 127;
    const int hw = idx & 1023;
    xT[(size_t)((b << 10) + hw) * CIN + c] = (h16_t)x[idx];
}

// ---------------------------------------------------------------------------
// BN statistics: one block per channel, reduce over all NPOS positions.
// ---------------------------------------------------------------------------
__global__ void __launch_bounds__(256)
bn_stats(const float* __restrict__ x, float* __restrict__ mean, float* __restrict__ var) {
    __shared__ float ss[256], ss2[256];
    const int m = blockIdx.x;
    float s = 0.0f, s2 = 0.0f;
    for (int n = threadIdx.x; n < NPOS; n += 256) {
        const float v = x[(size_t)n * CO + m];
        s += v; s2 += v * v;
    }
    ss[threadIdx.x] = s; ss2[threadIdx.x] = s2;
    __syncthreads();
    for (int st = 128; st > 0; st >>= 1) {
        if (threadIdx.x < st) {
            ss[threadIdx.x]  += ss[threadIdx.x + st];
            ss2[threadIdx.x] += ss2[threadIdx.x + st];
        }
        __syncthreads();
    }
    if (threadIdx.x == 0) {
        const float mu = ss[0] / (float)NPOS;
        mean[m] = mu;
        var[m]  = ss2[0] / (float)NPOS - mu * mu;
    }
}

__global__ void bn_apply(float* __restrict__ x,
                         const float* __restrict__ mean, const float* __restrict__ var,
                         const float* __restrict__ g, const float* __restrict__ b) {
    const int idx = blockIdx.x * 256 + threadIdx.x;     // over NPOS*CO
    const int m = idx & 255;
    x[idx] = g[m] * (x[idx] - mean[m]) * rsqrtf(var[m] + EPSV) + b[m];
}

// BN + relu, f32 in -> f16 out (feeds the next GEMM directly)
__global__ void bn_apply_out16(const float* __restrict__ x, h16_t* __restrict__ y,
                               const float* __restrict__ mean, const float* __restrict__ var,
                               const float* __restrict__ g, const float* __restrict__ b) {
    const int idx = blockIdx.x * 256 + threadIdx.x;
    const int m = idx & 255;
    float v = g[m] * (x[idx] - mean[m]) * rsqrtf(var[m] + EPSV) + b[m];
    y[idx] = (h16_t)fmaxf(v, 0.0f);
}

// ---------------------------------------------------------------------------
// Channel attention
// ---------------------------------------------------------------------------
__global__ void __launch_bounds__(256)
chan_pool(const float* __restrict__ x, float* __restrict__ avgp, float* __restrict__ maxp) {
    const int b = blockIdx.x, m = threadIdx.x;
    float s = 0.0f, mx = -3.4e38f;
    for (int p = 0; p < HW; ++p) {
        const float v = x[(size_t)((b << 10) + p) * CO + m];
        s += v; mx = fmaxf(mx, v);
    }
    avgp[b * CO + m] = s * (1.0f / (float)HW);
    maxp[b * CO + m] = mx;
}

__global__ void __launch_bounds__(256)
ca_mlp(const float* __restrict__ avgp, const float* __restrict__ maxp,
       const float* __restrict__ w1, const float* __restrict__ w2,
       float* __restrict__ ch_att) {
    __shared__ float sa[CO], sm[CO], hh[16];
    const int b = blockIdx.x, t = threadIdx.x;
    sa[t] = avgp[b * CO + t];
    sm[t] = maxp[b * CO + t];
    __syncthreads();
    if (t < 16) {
        float da = 0.0f, dm = 0.0f;
        for (int j = 0; j < CO; ++j) {
            const float w = w1[t * CO + j];
            da += w * sa[j]; dm += w * sm[j];
        }
        hh[t] = fmaxf(da, 0.0f) + fmaxf(dm, 0.0f);   // relu(avg)+relu(max)
    }
    __syncthreads();
    float o = 0.0f;
    for (int h = 0; h < 16; ++h) o += w2[t * 16 + h] * hh[h];
    ch_att[b * CO + t] = sigmoidf_(o);
}

__global__ void apply_ca(float* __restrict__ x, const float* __restrict__ ch_att) {
    const int idx = blockIdx.x * 256 + threadIdx.x;     // over NPOS*CO
    const int n = idx >> 8;
    const int b = n >> 10;
    x[idx] *= ch_att[(b << 8) + (idx & 255)];
}

// ---------------------------------------------------------------------------
// Spatial attention
// ---------------------------------------------------------------------------
__global__ void __launch_bounds__(256)
spat_pool(const float* __restrict__ x, float* __restrict__ sp) {
    const int wave = threadIdx.x >> 5, lane = threadIdx.x & 31;
    const int n = blockIdx.x * 8 + wave;
    const float* p = x + (size_t)n * CO;
    float s = 0.0f, mx = -3.4e38f;
    #pragma unroll
    for (int j = 0; j < 8; ++j) {
        const float v = p[lane + 32 * j];
        s += v; mx = fmaxf(mx, v);
    }
    for (int st = 16; st >= 1; st >>= 1) {
        s  += __shfl_xor(s, st, 32);
        mx  = fmaxf(mx, __shfl_xor(mx, st, 32));
    }
    if (lane == 0) { sp[n * 2] = s * (1.0f / (float)CO); sp[n * 2 + 1] = mx; }
}

__global__ void sa_conv(const float* __restrict__ sp, const float* __restrict__ w,
                        const float* __restrict__ bias, float* __restrict__ satt) {
    const int n = blockIdx.x * 256 + threadIdx.x;
    const int b = n >> 10, hw = n & 1023, h = hw >> 5, ww = hw & 31;
    float acc = bias[0];
    for (int dh = 0; dh < 7; ++dh) {
        const int hs = h + dh - 3;
        if (hs < 0 || hs >= HH) continue;
        for (int dw = 0; dw < 7; ++dw) {
            const int ws = ww + dw - 3;
            if (ws < 0 || ws >= WW2) continue;
            const int base = ((b << 10) + (hs << 5) + ws) * 2;
            acc += sp[base]     * w[dh * 7 + dw];
            acc += sp[base + 1] * w[49 + dh * 7 + dw];
        }
    }
    satt[n] = sigmoidf_(acc);
}

// x *= satt (in place, f32) and co-write the f16 mirror (xf for the GEMMs)
__global__ void apply_sa(float* __restrict__ x, const float* __restrict__ satt,
                         h16_t* __restrict__ y16) {
    const int idx = blockIdx.x * 256 + threadIdx.x;
    const float v = x[idx] * satt[idx >> 8];
    x[idx] = v;
    y16[idx] = (h16_t)v;
}

// ---------------------------------------------------------------------------
// Softmax over 64 columns, wave32 per row; writes f32 + f16 mirror.
// ---------------------------------------------------------------------------
__global__ void __launch_bounds__(256)
softmax64(float* __restrict__ a, h16_t* __restrict__ ah) {
    const int wave = threadIdx.x >> 5, lane = threadIdx.x & 31;
    const size_t row = (size_t)(blockIdx.x * 8 + wave) * MEMN;
    float* p = a + row;
    float v0 = p[lane], v1 = p[lane + 32];
    float mx = fmaxf(v0, v1);
    for (int st = 16; st >= 1; st >>= 1) mx = fmaxf(mx, __shfl_xor(mx, st, 32));
    const float e0 = __expf(v0 - mx), e1 = __expf(v1 - mx);
    float sm = e0 + e1;
    for (int st = 16; st >= 1; st >>= 1) sm += __shfl_xor(sm, st, 32);
    const float inv = 1.0f / sm;
    const float r0 = e0 * inv, r1 = e1 * inv;
    p[lane] = r0; p[lane + 32] = r1;
    ah[row + lane] = (h16_t)r0; ah[row + lane + 32] = (h16_t)r1;
}

// ---------------------------------------------------------------------------
// mo = gate*retrieved + (1-gate)*xf   (gate overwritten in place)
// ---------------------------------------------------------------------------
__global__ void mo_combine(float* __restrict__ gate, const float* __restrict__ retr,
                           const float* __restrict__ xf) {
    const int idx = blockIdx.x * 256 + threadIdx.x;
    const float g = gate[idx];
    gate[idx] = g * retr[idx] + (1.0f - g) * xf[idx];
}

// ---------------------------------------------------------------------------
// Final: spike = (0.1*mo >= 1), out = relu(spike + identity), NCHW output.
// ---------------------------------------------------------------------------
__global__ void finalize(float* __restrict__ out, const float* __restrict__ mo,
                         const float* __restrict__ idT) {
    const int idx = blockIdx.x * 256 + threadIdx.x;     // NCHW order, B*CO*HW
    const int b  = idx >> 18;                           // CO*HW = 262144
    const int rem = idx & 262143;
    const int m  = rem >> 10;
    const int hw = rem & 1023;
    const size_t nm = (size_t)((b << 10) + hw) * CO + m;
    const float s = (0.1f * mo[nm] >= 1.0f) ? 1.0f : 0.0f;
    out[idx] = fmaxf(s + idT[nm], 0.0f);
}

// ---------------------------------------------------------------------------
extern "C" void kernel_launch(void* const* d_in, const int* in_sizes, int n_in,
                              void* d_out, int out_size, void* d_ws, size_t ws_size,
                              hipStream_t stream) {
    const float* x       = (const float*)d_in[0];
    const float* conv1_w = (const float*)d_in[1];
    const float* conv1_b = (const float*)d_in[2];
    const float* bn1_g   = (const float*)d_in[3];
    const float* bn1_b   = (const float*)d_in[4];
    const float* conv2_w = (const float*)d_in[5];
    const float* conv2_b = (const float*)d_in[6];
    const float* bn2_g   = (const float*)d_in[7];
    const float* bn2_b   = (const float*)d_in[8];
    const float* ca_w1   = (const float*)d_in[9];
    const float* ca_w2   = (const float*)d_in[10];
    const float* sa_w    = (const float*)d_in[11];
    const float* sa_b    = (const float*)d_in[12];
    const float* mem     = (const float*)d_in[13];
    const float* mem_k   = (const float*)d_in[14];
    const float* ctrl_w1 = (const float*)d_in[15];
    const float* ctrl_b1 = (const float*)d_in[16];
    const float* ctrl_w2 = (const float*)d_in[17];
    const float* ctrl_b2 = (const float*)d_in[18];
    const float* gate_w  = (const float*)d_in[19];
    const float* gate_b  = (const float*)d_in[20];
    const float* sc_w    = (const float*)d_in[21];
    const float* sc_g    = (const float*)d_in[22];
    const float* sc_b    = (const float*)d_in[23];
    float* out = (float*)d_out;

    // -------- workspace layout --------
    float* ws = (float*)d_ws;
    float* bufI = ws;                            // NPOS*CO f32 (identity)
    float* bufA = bufI + (size_t)NPOS * CO;      // NPOS*CO f32
    float* bufB = bufA + (size_t)NPOS * CO;      // NPOS*CO f32 (xf)
    float* bufC = bufB + (size_t)NPOS * CO;      // NPOS*CO f32 (logits/gate/mo)
    float* attn = bufC;                          // overlap: NPOS*MEMN, lifetime
                                                 // ends before gate writes bufC
    float* bnm  = bufC + (size_t)NPOS * CO;      // 256
    float* bnv  = bnm + CO;
    float* avgp = bnv + CO;                      // BB*CO
    float* maxp = avgp + BB * CO;
    float* chat = maxp + BB * CO;
    float* sp   = chat + BB * CO;                // NPOS*2
    float* satt = sp + NPOS * 2;                 // NPOS
    // f16 region
    h16_t* hbase = (h16_t*)(satt + NPOS);
    h16_t* xTh   = hbase;                        // NPOS*CIN
    h16_t* bufAh = xTh   + (size_t)NPOS * CIN;   // NPOS*CO
    h16_t* bufBh = bufAh + (size_t)NPOS * CO;    // NPOS*CO (xf)
    h16_t* bufCh = bufBh + (size_t)NPOS * CO;    // NPOS*CO (q)
    h16_t* attnh = bufCh + (size_t)NPOS * CO;    // NPOS*MEMN
    h16_t* w1h   = attnh + (size_t)NPOS * MEMN;  // CO*9*CIN   = 294912
    h16_t* w2h   = w1h + CO * 9 * CIN;           // CO*9*CO    = 589824
    h16_t* scwh  = w2h + CO * 9 * CO;            // CO*CIN
    h16_t* c1h   = scwh + CO * CIN;              // CO*CO
    h16_t* c2h   = c1h + CO * CO;                // CO*CO
    h16_t* mkh   = c2h + CO * CO;                // MEMN*CO
    h16_t* memTh = mkh + MEMN * CO;              // CO*MEMN
    h16_t* gwh   = memTh + CO * MEMN;            // CO*2*CO

    const dim3 blk(256);
    const dim3 gEw(NPOS * CO / 256);             // elementwise over N x 256
    const dim3 gA(CO / 128, NPOS / 128);         // big GEMMs: block 128x128
    const dim3 gB(1, NPOS / 128);                // logits: block 64x128

    // -------- stage all GEMM operands as f16 --------
    nchw_to_nhwc16<<<BB * CIN * HW / 256, blk, 0, stream>>>(x, xTh);
    repack_w3x3<<<dim3(9 * CIN / 128, CO), dim3(128), 0, stream>>>(conv1_w, w1h, 7);
    repack_w3x3<<<dim3(9 * CO / 128, CO), dim3(128), 0, stream>>>(conv2_w, w2h, 8);
    cvtf16<<<CO * CIN / 256, blk, 0, stream>>>(sc_w, scwh);
    cvtf16<<<CO * CO / 256, blk, 0, stream>>>(ctrl_w1, c1h);
    cvtf16<<<CO * CO / 256, blk, 0, stream>>>(ctrl_w2, c2h);
    cvtf16<<<MEMN * CO / 256, blk, 0, stream>>>(mem_k, mkh);
    cvtf16<<<CO * 2 * CO / 256, blk, 0, stream>>>(gate_w, gwh);
    repack_memT<<<CO * MEMN / 256, blk, 0, stream>>>(mem, memTh);

    // 1) identity = BN(1x1 conv)
    wmma_gemm<1,7,0,0,0,0,0, 4,2,2,4><<<gA, blk, 0, stream>>>(
        bufI, nullptr, xTh, nullptr, scwh, nullptr, CIN, CO, CIN, 0, CIN, CIN, 1.0f);
    bn_stats<<<CO, blk, 0, stream>>>(bufI, bnm, bnv);
    bn_apply<<<gEw, blk, 0, stream>>>(bufI, bnm, bnv, sc_g, sc_b);

    // 2) conv1 (3x3 implicit GEMM, K = 9*128) -> f32, then BN+relu -> f16
    wmma_gemm<9,7,0,1,0,0,0, 4,2,2,4><<<gA, blk, 0, stream>>>(
        bufA, nullptr, xTh, nullptr, w1h, conv1_b, 9 * CIN, CO, 0, 0, 0, 9 * CIN, 1.0f);
    bn_stats<<<CO, blk, 0, stream>>>(bufA, bnm, bnv);
    bn_apply_out16<<<gEw, blk, 0, stream>>>(bufA, bufAh, bnm, bnv, bn1_g, bn1_b);

    // 3) conv2 (3x3, K = 9*256) + BN (f32, feeds pooling chain)
    wmma_gemm<9,8,0,1,0,0,0, 4,2,2,4><<<gA, blk, 0, stream>>>(
        bufB, nullptr, bufAh, nullptr, w2h, conv2_b, 9 * CO, CO, 0, 0, 0, 9 * CO, 1.0f);
    bn_stats<<<CO, blk, 0, stream>>>(bufB, bnm, bnv);
    bn_apply<<<gEw, blk, 0, stream>>>(bufB, bnm, bnv, bn2_g, bn2_b);

    // 4) channel attention
    chan_pool<<<BB, blk, 0, stream>>>(bufB, avgp, maxp);
    ca_mlp<<<BB, blk, 0, stream>>>(avgp, maxp, ca_w1, ca_w2, chat);
    apply_ca<<<gEw, blk, 0, stream>>>(bufB, chat);

    // 5) spatial attention; bufB = xf (f32) + bufBh (f16)
    spat_pool<<<NPOS / 8, blk, 0, stream>>>(bufB, sp);
    sa_conv<<<NPOS / 256, blk, 0, stream>>>(sp, sa_w, sa_b, satt);
    apply_sa<<<gEw, blk, 0, stream>>>(bufB, satt, bufBh);

    // 6) memory module
    //    h1 = relu(xf @ ctrl_w1^T + b1) -> f16 only
    wmma_gemm<1,7,1,1,0,0,1, 4,2,2,4><<<gA, blk, 0, stream>>>(
        nullptr, bufAh, bufBh, nullptr, c1h, ctrl_b1, CO, CO, CO, 0, CO, CO, 1.0f);
    //    q = h1 @ ctrl_w2^T + b2 -> f16 only
    wmma_gemm<1,7,0,1,0,0,1, 4,2,2,4><<<gA, blk, 0, stream>>>(
        nullptr, bufCh, bufAh, nullptr, c2h, ctrl_b2, CO, CO, CO, 0, CO, CO, 1.0f);
    //    logits = (q @ mem_keys^T) / 16   (M=64: block 64x128, 2x4 waves)
    wmma_gemm<1,7,0,0,0,1,0, 2,4,2,2><<<gB, blk, 0, stream>>>(
        attn, nullptr, bufCh, nullptr, mkh, nullptr, CO, MEMN, CO, 0, CO, CO,
        1.0f / 16.0f);
    softmax64<<<NPOS / 8, blk, 0, stream>>>(attn, attnh);
    //    retrieved = attn @ mem (memT row-major, K=64) -> f32 + f16
    wmma_gemm<1,7,0,0,0,0,2, 4,2,2,4><<<gA, blk, 0, stream>>>(
        bufA, bufAh, attnh, nullptr, memTh, nullptr, MEMN, CO, MEMN, 0, MEMN, MEMN, 1.0f);
    //    gate = sigmoid([xf | retrieved] @ gate_w^T + gate_b), K=512 split 256
    //    (overwrites bufC after attn's lifetime has ended)
    wmma_gemm<1,7,2,1,1,0,0, 4,2,2,4><<<gA, blk, 0, stream>>>(
        bufC, nullptr, bufBh, bufAh, gwh, gate_b, 2 * CO, CO, CO, CO, CO, 2 * CO, 1.0f);
    //    mo = gate*retrieved + (1-gate)*xf  (in place over gate -> bufC)
    mo_combine<<<gEw, blk, 0, stream>>>(bufC, bufA, bufB);

    // 7) spike + residual + relu, back to NCHW
    finalize<<<gEw, blk, 0, stream>>>(out, bufC, bufI);
}
